// MultiHeadSelfAttention_18193481465897
// MI455X (gfx1250) — compile-verified, matrix operand order checked by reading, not run on hardware
//
#include <hip/hip_runtime.h>

// ---------------------------------------------------------------------------
// MI455X (gfx1250) fused MHA: bf16 cast -> QKV proj (TDM + WMMA bf16) ->
// rmsnorm+rope -> flash attention (WMMA bf16) -> gated merge -> out proj.
// ---------------------------------------------------------------------------

typedef __attribute__((ext_vector_type(16))) __bf16 v16bf;
typedef __attribute__((ext_vector_type(8)))  float  v8f;
typedef __attribute__((ext_vector_type(4)))  float  f32x4;
typedef __attribute__((ext_vector_type(4)))  unsigned int u32x4;
typedef __attribute__((ext_vector_type(4)))  int    i32x4;
typedef __attribute__((ext_vector_type(8)))  int    i32x8;

#define D_MODEL   1024
#define NUM_HEADS 16
#define HEAD_DIM  64
#define BSZ       4
#define SEQ       2048
#define NTOK      (BSZ * SEQ)          // 8192

union Frag32B { v16bf v; u32x4 u[2]; };

__device__ __forceinline__ v8f wmma_bf16(v16bf a, v16bf b, v8f c) {
    return __builtin_amdgcn_wmma_f32_16x16x32_bf16(
        /*neg_a=*/false, a, /*neg_b=*/false, b,
        /*c_mod=*/(short)0, c, /*reuse_a=*/false, /*reuse_b=*/false);
}

// Load one 16x32 bf16 A/B fragment. `row_base` points to this lane's row
// (K-contiguous). Lane l<16 takes K chunks [0..7],[16..23]; lane l>=16 takes
// [8..15],[24..31] (per CDNA5 ISA 16-bit matrix layout).
__device__ __forceinline__ v16bf load_frag(const __bf16* row_base, int lane, int kchunk) {
    int e0 = kchunk + ((lane & 16) ? 8 : 0);
    Frag32B f;
    f.u[0] = *(const u32x4*)(row_base + e0);
    f.u[1] = *(const u32x4*)(row_base + e0 + 16);
    return f.v;
}

// ---------------------------------------------------------------------------
// TDM: DMA one 128-row x 32-col bf16 tile (row-major, dim0 = K contiguous)
// from global into LDS at `lds_addr`. Issued once per wave; TENSORcnt-tracked.
// D# layout per CDNA5 ISA 8.3/8.4 (2D tensor: groups 2/3 zero).
// ---------------------------------------------------------------------------
__device__ __forceinline__ void tdm_load_tile_bf16(const __bf16* gsrc, unsigned lds_addr,
                                                   unsigned tensor_rows, unsigned tensor_k) {
    unsigned long long ga = (unsigned long long)(uintptr_t)gsrc;
    u32x4 g0;
    g0[0] = 1u;                                           // count=1 (valid user D#)
    g0[1] = lds_addr;                                     // LDS byte address
    g0[2] = (unsigned)(ga & 0xFFFFFFFFu);                 // global_addr[31:0]
    g0[3] = (unsigned)((ga >> 32) & 0x01FFFFFFu) | (2u << 30);  // [56:32] | type=2
    i32x8 g1;
    g1[0] = (int)(1u << 16);                              // data_size=1 (2 bytes)
    g1[1] = (int)((tensor_k & 0xFFFFu) << 16);            // tensor_dim0[15:0]
    g1[2] = (int)(((tensor_k >> 16) & 0xFFFFu) |
                  ((tensor_rows & 0xFFFFu) << 16));       // dim0[31:16] | dim1[15:0]
    g1[3] = (int)(((tensor_rows >> 16) & 0xFFFFu) |
                  (32u << 16));                           // dim1[31:16] | tile_dim0=32
    g1[4] = 128;                                          // tile_dim1=128, tile_dim2=0
    g1[5] = (int)tensor_k;                                // tensor_dim0_stride (elems)
    g1[6] = 0;
    g1[7] = 0;
    i32x4 gz = {};
#if defined(__clang_major__) && (__clang_major__ >= 23)
    i32x8 gz8 = {};
    __builtin_amdgcn_tensor_load_to_lds(g0, g1, gz, gz, gz8, 0);
#else
    __builtin_amdgcn_tensor_load_to_lds(g0, g1, gz, gz, 0);
#endif
}

// ---------------------------------------------------------------------------
// GEMM: C[M,N] = A[M,K] * B[N,K]^T + bias[N]   (A,B bf16, f32 accum/out)
// 128 threads = 4 waves; block tile 128x128; wave tile 64x64; BK=32.
// A/B tiles staged by TDM, double-buffered against the WMMA loop.
// ---------------------------------------------------------------------------
__global__ __launch_bounds__(128)
void gemm_bias_kernel(const __bf16* __restrict__ A, const __bf16* __restrict__ B,
                      const float* __restrict__ bias, float* __restrict__ C,
                      int M, int N, int K) {
    __shared__ __bf16 As[2][128 * 32];
    __shared__ __bf16 Bs[2][128 * 32];

    const int tid  = threadIdx.x;
    const int lane = tid & 31;
    const int wave = tid >> 5;
    const int wm   = wave >> 1;            // 0..1
    const int wn   = wave & 1;             // 0..1
    const int bm   = blockIdx.y * 128;
    const int bn   = blockIdx.x * 128;

    const unsigned as_addr[2] = { (unsigned)(uintptr_t)(void*)&As[0][0],
                                  (unsigned)(uintptr_t)(void*)&As[1][0] };
    const unsigned bs_addr[2] = { (unsigned)(uintptr_t)(void*)&Bs[0][0],
                                  (unsigned)(uintptr_t)(void*)&Bs[1][0] };

    v8f acc[4][4] = {};

    const int nsteps = K / 32;
    if (wave == 0) {
        tdm_load_tile_bf16(A + (size_t)bm * K, as_addr[0], (unsigned)M, (unsigned)K);
        tdm_load_tile_bf16(B + (size_t)bn * K, bs_addr[0], (unsigned)N, (unsigned)K);
    }

    int buf = 0;
    for (int i = 0; i < nsteps; ++i) {
        if (wave == 0) {
            if (i + 1 < nsteps) {
                // prefetch next stage into the other buffer, then retire stage i
                tdm_load_tile_bf16(A + (size_t)bm * K + (i + 1) * 32,
                                   as_addr[buf ^ 1], (unsigned)M, (unsigned)K);
                tdm_load_tile_bf16(B + (size_t)bn * K + (i + 1) * 32,
                                   bs_addr[buf ^ 1], (unsigned)N, (unsigned)K);
                __builtin_amdgcn_s_wait_tensorcnt(2);
            } else {
                __builtin_amdgcn_s_wait_tensorcnt(0);
            }
        }
        __syncthreads();

        const __bf16* Ab = &As[buf][0];
        const __bf16* Bb = &Bs[buf][0];
        v16bf af[4], bf[4];
#pragma unroll
        for (int ii = 0; ii < 4; ++ii)
            af[ii] = load_frag(Ab + (wm * 64 + ii * 16 + (lane & 15)) * 32, lane, 0);
#pragma unroll
        for (int j = 0; j < 4; ++j)
            bf[j] = load_frag(Bb + (wn * 64 + j * 16 + (lane & 15)) * 32, lane, 0);

#pragma unroll
        for (int ii = 0; ii < 4; ++ii)
#pragma unroll
            for (int j = 0; j < 4; ++j)
                acc[ii][j] = wmma_bf16(af[ii], bf[j], acc[ii][j]);

        __syncthreads();
        buf ^= 1;
    }

    // Epilogue: C-matrix layout -> lane l covers col (l&15), rows r + 8*(l>>4).
#pragma unroll
    for (int i = 0; i < 4; ++i) {
#pragma unroll
        for (int j = 0; j < 4; ++j) {
            int col  = bn + wn * 64 + j * 16 + (lane & 15);
            float bv = bias[col];
            int rbase = bm + wm * 64 + i * 16 + ((lane >> 4) << 3);
#pragma unroll
            for (int r = 0; r < 8; ++r)
                C[(size_t)(rbase + r) * N + col] = acc[i][j][r] + bv;
        }
    }
}

// ---------------------------------------------------------------------------
// f32 -> bf16 cast, 4 elems/thread, 8-byte packed stores.
// ---------------------------------------------------------------------------
__global__ __launch_bounds__(256)
void cast_bf16_kernel(const float* __restrict__ src, __bf16* __restrict__ dst) {
    size_t i = ((size_t)blockIdx.x * 256 + threadIdx.x) * 4;
    f32x4 f = *(const f32x4*)(src + i);
    union { __bf16 h[4]; unsigned long long u; } t;
    t.h[0] = (__bf16)f.x; t.h[1] = (__bf16)f.y;
    t.h[2] = (__bf16)f.z; t.h[3] = (__bf16)f.w;
    *(unsigned long long*)(dst + i) = t.u;
}

// ---------------------------------------------------------------------------
// RMS-norm over d_model + RoPE, writes bf16 [B, H, S, D].
// One block (256 threads) per token; each thread owns 4 channels (2 RoPE pairs).
// ---------------------------------------------------------------------------
__global__ __launch_bounds__(256)
void rmsnorm_rope_kernel(const float* __restrict__ pre, const float* __restrict__ w,
                         const int* __restrict__ start_pos, __bf16* __restrict__ out) {
    __shared__ float red[256];
    const int t   = blockIdx.x;          // token
    const int b   = t >> 11;             // /SEQ
    const int s   = t & (SEQ - 1);
    const int tid = threadIdx.x;

    f32x4 xv = ((const f32x4*)(pre + (size_t)t * D_MODEL))[tid];
    red[tid] = xv.x * xv.x + xv.y * xv.y + xv.z * xv.z + xv.w * xv.w;
    __syncthreads();
#pragma unroll
    for (int o = 128; o > 0; o >>= 1) {
        if (tid < o) red[tid] += red[tid + o];
        __syncthreads();
    }
    const float scale = rsqrtf(red[0] * (1.0f / D_MODEL) + 1.1920929e-07f);

    f32x4 wv = ((const f32x4*)w)[tid];
    float y0 = xv.x * scale * wv.x;
    float y1 = xv.y * scale * wv.y;
    float y2 = xv.z * scale * wv.z;
    float y3 = xv.w * scale * wv.w;

    const int c = tid * 4;
    const int h = c >> 6;
    const int d = c & 63;
    const float pos = (float)(start_pos[0] + s);

    float inv0 = __powf(10000.0f, -((float)d) / 64.0f);
    float inv1 = __powf(10000.0f, -((float)(d + 2)) / 64.0f);
    float s0, c0, s1, c1;
    __sincosf(pos * inv0, &s0, &c0);
    __sincosf(pos * inv1, &s1, &c1);

    float o0 = y0 * c0 - y1 * s0;
    float o1 = y0 * s0 + y1 * c0;
    float o2 = y2 * c1 - y3 * s1;
    float o3 = y2 * s1 + y3 * c1;

    __bf16* op = out + (((size_t)(b * NUM_HEADS + h) * SEQ + s) * HEAD_DIM + d);
    op[0] = (__bf16)o0;
    op[1] = (__bf16)o1;
    op[2] = (__bf16)o2;
    op[3] = (__bf16)o3;
}

// ---------------------------------------------------------------------------
// V pack + transpose: f32 [B*S, D_MODEL] -> bf16 [B, H, D, S]
// ---------------------------------------------------------------------------
__global__ __launch_bounds__(256)
void pack_vt_kernel(const float* __restrict__ v, __bf16* __restrict__ vt) {
    size_t i = (size_t)blockIdx.x * 256 + threadIdx.x;   // over NTOK*D_MODEL
    int tkn = (int)(i >> 10);
    int c   = (int)(i & 1023);
    int b = tkn >> 11, s = tkn & (SEQ - 1);
    int h = c >> 6,    d = c & 63;
    vt[((size_t)(b * NUM_HEADS + h) * HEAD_DIM + d) * SEQ + s] = (__bf16)v[i];
}

// ---------------------------------------------------------------------------
// Flash attention (non-causal), per block: (64 queries, one (b,h)); 4 waves,
// each wave owns 16 query rows. Online softmax; P goes through LDS to convert
// C-layout -> A-layout for the P@V WMMA. Gate applied at the end; ctx is bf16.
// ---------------------------------------------------------------------------
__global__ __launch_bounds__(128)
void attn_kernel(const __bf16* __restrict__ q, const __bf16* __restrict__ k,
                 const __bf16* __restrict__ vt, const float* __restrict__ gate_logits,
                 __bf16* __restrict__ ctx) {
    __shared__ __bf16 P[4][16 * 64];

    const int lane = threadIdx.x & 31;
    const int wave = threadIdx.x >> 5;
    const int b = blockIdx.z;
    const int h = blockIdx.y;
    const int m0 = blockIdx.x * 64 + wave * 16;

    const size_t head_off = (size_t)(b * NUM_HEADS + h) * SEQ * HEAD_DIM;
    const __bf16* qh  = q  + head_off;
    const __bf16* kh  = k  + head_off;
    const __bf16* vth = vt + head_off;   // [HEAD_DIM][SEQ] for this head

    const __bf16* qrow = qh + (size_t)(m0 + (lane & 15)) * HEAD_DIM;
    v16bf qf0 = load_frag(qrow, lane, 0);
    v16bf qf1 = load_frag(qrow, lane, 32);

    v8f o[4] = {};
    float mrun[8], lrun[8];
#pragma unroll
    for (int r = 0; r < 8; ++r) { mrun[r] = -__builtin_inff(); lrun[r] = 0.0f; }

    for (int kt = 0; kt < SEQ; kt += 64) {
        // ---- S = scale * Q K^T for 64 keys (4 x 16x16 tiles) ----
        v8f sacc[4];
#pragma unroll
        for (int j = 0; j < 4; ++j) {
            const __bf16* krow = kh + (size_t)(kt + j * 16 + (lane & 15)) * HEAD_DIM;
            v16bf kf0 = load_frag(krow, lane, 0);
            v16bf kf1 = load_frag(krow, lane, 32);
            v8f z = {};
            z = wmma_bf16(qf0, kf0, z);
            z = wmma_bf16(qf1, kf1, z);
            sacc[j] = z;
        }
#pragma unroll
        for (int j = 0; j < 4; ++j)
#pragma unroll
            for (int r = 0; r < 8; ++r) sacc[j][r] *= 0.125f;   // 1/sqrt(64)

        // ---- online softmax update (rows spread over 16-lane groups) ----
        float mnew[8], alpha[8];
#pragma unroll
        for (int r = 0; r < 8; ++r) {
            float t = fmaxf(fmaxf(sacc[0][r], sacc[1][r]), fmaxf(sacc[2][r], sacc[3][r]));
#pragma unroll
            for (int off = 8; off >= 1; off >>= 1) t = fmaxf(t, __shfl_xor(t, off, 32));
            mnew[r]  = fmaxf(mrun[r], t);
            alpha[r] = __expf(mrun[r] - mnew[r]);
        }
#pragma unroll
        for (int r = 0; r < 8; ++r) {
            float rs = 0.0f;
#pragma unroll
            for (int j = 0; j < 4; ++j) {
                float p = __expf(sacc[j][r] - mnew[r]);
                sacc[j][r] = p;
                rs += p;
            }
#pragma unroll
            for (int off = 8; off >= 1; off >>= 1) rs += __shfl_xor(rs, off, 32);
            lrun[r] = lrun[r] * alpha[r] + rs;
            mrun[r] = mnew[r];
        }
#pragma unroll
        for (int jd = 0; jd < 4; ++jd)
#pragma unroll
            for (int r = 0; r < 8; ++r) o[jd][r] *= alpha[r];

        // ---- P: C-layout -> LDS (16x64 bf16) -> A-layout fragments ----
        __bf16* pb = &P[wave][0];
        const int colb = lane & 15;
        const int rowo = (lane >> 4) * 8;
#pragma unroll
        for (int j = 0; j < 4; ++j)
#pragma unroll
            for (int r = 0; r < 8; ++r)
                pb[(rowo + r) * 64 + j * 16 + colb] = (__bf16)sacc[j][r];
        asm volatile("s_wait_dscnt 0" ::: "memory");

        const __bf16* prow = pb + (lane & 15) * 64;
        v16bf pf0 = load_frag(prow, lane, 0);
        v16bf pf1 = load_frag(prow, lane, 32);

        // ---- O += P @ V  (V^T rows are d-contiguous keys) ----
#pragma unroll
        for (int jd = 0; jd < 4; ++jd) {
            const __bf16* vrow = vth + (size_t)(jd * 16 + (lane & 15)) * SEQ + kt;
            v16bf vf0 = load_frag(vrow, lane, 0);
            v16bf vf1 = load_frag(vrow, lane, 32);
            o[jd] = wmma_bf16(pf0, vf0, o[jd]);
            o[jd] = wmma_bf16(pf1, vf1, o[jd]);
        }
    }

    const float gl = gate_logits[h];
    const float gate = 1.0f / (1.0f + __expf(-gl));
#pragma unroll
    for (int jd = 0; jd < 4; ++jd) {
        int dcol = h * HEAD_DIM + jd * 16 + (lane & 15);
#pragma unroll
        for (int r = 0; r < 8; ++r) {
            int row = m0 + ((lane >> 4) << 3) + r;
            ctx[((size_t)(b * SEQ + row)) * D_MODEL + dcol] =
                (__bf16)(o[jd][r] / lrun[r] * gate);
        }
    }
}

// ---------------------------------------------------------------------------
// Host launcher
// ---------------------------------------------------------------------------
extern "C" void kernel_launch(void* const* d_in, const int* in_sizes, int n_in,
                              void* d_out, int out_size, void* d_ws, size_t ws_size,
                              hipStream_t stream) {
    const float* x   = (const float*)d_in[0];
    const float* Wq  = (const float*)d_in[1];
    const float* bq  = (const float*)d_in[2];
    const float* Wk  = (const float*)d_in[3];
    const float* bk  = (const float*)d_in[4];
    const float* Wv  = (const float*)d_in[5];
    const float* bv  = (const float*)d_in[6];
    const float* Wo  = (const float*)d_in[7];
    const float* bo  = (const float*)d_in[8];
    const float* qnw = (const float*)d_in[9];
    const float* knw = (const float*)d_in[10];
    const float* gl  = (const float*)d_in[11];
    const int*   sp  = (const int*)d_in[12];
    float* out = (float*)d_out;

    char* ws = (char*)d_ws;
    const size_t BF_TOK = (size_t)NTOK * D_MODEL * sizeof(__bf16);   // 16.8 MB
    const size_t BF_W   = (size_t)D_MODEL * D_MODEL * sizeof(__bf16); // 2.1 MB
    const size_t F32_TOK = (size_t)NTOK * D_MODEL * sizeof(float);   // 33.5 MB

    size_t off = 0;
    __bf16* xb   = (__bf16*)(ws + off); off += BF_TOK;
    __bf16* Wq_b = (__bf16*)(ws + off); off += BF_W;
    __bf16* Wk_b = (__bf16*)(ws + off); off += BF_W;
    __bf16* Wv_b = (__bf16*)(ws + off); off += BF_W;
    __bf16* Wo_b = (__bf16*)(ws + off); off += BF_W;
    float*  q_pre = (float*)(ws + off); off += F32_TOK;
    float*  k_pre = (float*)(ws + off); off += F32_TOK;
    float*  v_pre = (float*)(ws + off); off += F32_TOK;
    __bf16* q_bf  = (__bf16*)(ws + off); off += BF_TOK;
    __bf16* k_bf  = (__bf16*)(ws + off); off += BF_TOK;
    __bf16* vt_bf = (__bf16*)(ws + off); off += BF_TOK;
    __bf16* ctx_bf = (__bf16*)q_pre;   // q_pre dead after rmsnorm_rope -> reuse

    // bf16 casts (x once, 4 weight matrices)
    cast_bf16_kernel<<<(NTOK * D_MODEL) / 1024, 256, 0, stream>>>(x, xb);
    cast_bf16_kernel<<<(D_MODEL * D_MODEL) / 1024, 256, 0, stream>>>(Wq, Wq_b);
    cast_bf16_kernel<<<(D_MODEL * D_MODEL) / 1024, 256, 0, stream>>>(Wk, Wk_b);
    cast_bf16_kernel<<<(D_MODEL * D_MODEL) / 1024, 256, 0, stream>>>(Wv, Wv_b);
    cast_bf16_kernel<<<(D_MODEL * D_MODEL) / 1024, 256, 0, stream>>>(Wo, Wo_b);

    dim3 gg(D_MODEL / 128, NTOK / 128);   // (8, 64)
    gemm_bias_kernel<<<gg, 128, 0, stream>>>(xb, Wq_b, bq, q_pre, NTOK, D_MODEL, D_MODEL);
    gemm_bias_kernel<<<gg, 128, 0, stream>>>(xb, Wk_b, bk, k_pre, NTOK, D_MODEL, D_MODEL);
    gemm_bias_kernel<<<gg, 128, 0, stream>>>(xb, Wv_b, bv, v_pre, NTOK, D_MODEL, D_MODEL);

    rmsnorm_rope_kernel<<<NTOK, 256, 0, stream>>>(q_pre, qnw, sp, q_bf);
    rmsnorm_rope_kernel<<<NTOK, 256, 0, stream>>>(k_pre, knw, sp, k_bf);
    pack_vt_kernel<<<(NTOK * D_MODEL) / 256, 256, 0, stream>>>(v_pre, vt_bf);

    attn_kernel<<<dim3(SEQ / 64, NUM_HEADS, BSZ), 128, 0, stream>>>(q_bf, k_bf, vt_bf, gl, ctx_bf);

    gemm_bias_kernel<<<gg, 128, 0, stream>>>(ctx_bf, Wo_b, bo, out, NTOK, D_MODEL, D_MODEL);
}
